// MoELayer_90598040142495
// MI455X (gfx1250) — matrix operand order repaired and checked by reading, not compiled, not run on hardware
//
#include <hip/hip_runtime.h>
#include <hip/hip_bf16.h>
#include <math.h>

// ---------------------------------------------------------------------------
// Top-2 routed MoE for MI455X (gfx1250, wave32, WMMA).
// Routing in fp32; expert GEMMs bf16 WMMA w/ fp32 accumulate (~68.7 GF routed,
// ~170 MB traffic -> near roofline ridge; bf16 WMMA is the throughput path).
// GEMMs: 64x256 block tile (8 waves, each 64x32 -> 8 accumulators), so each
// B-fragment pair feeds 8 WMMAs (4x B reuse vs 16-row tiles). A staged in LDS
// via double-buffered global_load_async_to_lds_b128 (ASYNCcnt); B streamed
// from L2 (all expert weights fit in 192MB L2) with global_prefetch hints.
// ---------------------------------------------------------------------------

constexpr int kN = 4096;   // tokens
constexpr int kD = 1024;   // model dim
constexpr int kH = 2048;   // hidden dim
constexpr int kE = 8;      // experts
constexpr int ROWS_MAX = 2 * kN + kE * 64;  // 8704 (per-expert 64-row padding)
constexpr int KT = 128;    // K-chunk per LDS buffer (double-buffered)

typedef __attribute__((ext_vector_type(16))) __bf16        v16bf;
typedef __attribute__((ext_vector_type(8)))  float         v8f;
typedef __attribute__((ext_vector_type(4)))  unsigned int  v4u;
typedef __attribute__((ext_vector_type(2)))  unsigned int  v2u;

union FragBF {            // 32 bytes = 16 bf16 = one WMMA A/B operand per lane
    v16bf v;
    v4u   q[2];
};

__device__ __forceinline__ unsigned short f2bf(float f) {
    unsigned int u = __float_as_uint(f);
    unsigned int r = u + 0x7FFFu + ((u >> 16) & 1u);   // round-to-nearest-even
    if ((u & 0x7F800000u) == 0x7F800000u) r = u;       // keep Inf/NaN bits
    return (unsigned short)(r >> 16);
}

// Async global->LDS copy, 16B per lane, tracked by ASYNCcnt.
// Generic __shared__ addresses carry the LDS offset in the low 32 bits
// (flat aperture).
__device__ __forceinline__ void async_ld16(unsigned short* lds,
                                           const unsigned short* g) {
    unsigned laddr = (unsigned)(unsigned long long)lds;
    asm volatile("global_load_async_to_lds_b128 %0, %1, off"
                 :: "v"(laddr), "v"(g)
                 : "memory");
}
__device__ __forceinline__ void wait_async_keep4() {
    asm volatile("s_wait_asynccnt 0x4" ::: "memory");
}
__device__ __forceinline__ void wait_async_all() {
    asm volatile("s_wait_asynccnt 0x0" ::: "memory");
}

// --------------------------- setup / conversion ----------------------------

__global__ void zero_counts_kernel(int* counts) {
    if (threadIdx.x < kE) counts[threadIdx.x] = 0;
}

__global__ __launch_bounds__(256) void convert_x_kernel(
    const float* __restrict__ x, unsigned short* __restrict__ Xb) {
    int i = blockIdx.x * 256 + threadIdx.x;           // groups of 4 elements
    if (i >= kN * kD / 4) return;
    float4 f = ((const float4*)x)[i];
    unsigned short o[4] = { f2bf(f.x), f2bf(f.y), f2bf(f.z), f2bf(f.w) };
    v2u pk;
    pk.x = (unsigned int)o[0] | ((unsigned int)o[1] << 16);
    pk.y = (unsigned int)o[2] | ((unsigned int)o[3] << 16);
    ((v2u*)Xb)[i] = pk;
}

// src fp32 [E][R][C] -> dst bf16 [E][C][R]   (LDS-tiled transpose)
__global__ __launch_bounds__(256) void transpose_bf16_kernel(
    const float* __restrict__ src, unsigned short* __restrict__ dst,
    int R, int C) {
    __shared__ unsigned short tile[32][33];
    const int e  = blockIdx.z;
    const int c0 = blockIdx.x * 32;
    const int r0 = blockIdx.y * 32;
    const int tx = threadIdx.x & 31;
    const int ty = threadIdx.x >> 5;                  // 0..7
    const float* s = src + (size_t)e * R * C;
    for (int i = 0; i < 32; i += 8)
        tile[ty + i][tx] = f2bf(s[(size_t)(r0 + ty + i) * C + c0 + tx]);
    __syncthreads();
    unsigned short* d = dst + (size_t)e * R * C;
    for (int i = 0; i < 32; i += 8)
        d[(size_t)(c0 + ty + i) * R + r0 + tx] = tile[tx][ty + i];
}

// ------------------------------- gating ------------------------------------

__global__ __launch_bounds__(256) void gate_kernel(
    const float* __restrict__ x, const float* __restrict__ gW,
    const float* __restrict__ gb, float* __restrict__ gates,
    int* __restrict__ eidx, int* __restrict__ slots, int* __restrict__ counts) {
    const int n    = blockIdx.x * 8 + (threadIdx.x >> 5);  // one wave32/token
    const int lane = threadIdx.x & 31;
    float acc[kE];
#pragma unroll
    for (int e = 0; e < kE; ++e) acc[e] = 0.f;
    const float* xr = x + (size_t)n * kD;
    for (int d = lane; d < kD; d += 32) {
        float xv = xr[d];
        const float* g = gW + (size_t)d * kE;          // row-contiguous 32B
#pragma unroll
        for (int e = 0; e < kE; ++e) acc[e] += xv * g[e];
    }
#pragma unroll
    for (int e = 0; e < kE; ++e)
        for (int off = 16; off > 0; off >>= 1)
            acc[e] += __shfl_xor(acc[e], off, 32);
    float lg[kE], p[kE];
    float mx = -1e30f;
#pragma unroll
    for (int e = 0; e < kE; ++e) { lg[e] = acc[e] + gb[e]; mx = fmaxf(mx, lg[e]); }
    float s = 0.f;
#pragma unroll
    for (int e = 0; e < kE; ++e) { p[e] = __expf(lg[e] - mx); s += p[e]; }
    float inv = 1.f / s;
#pragma unroll
    for (int e = 0; e < kE; ++e) p[e] *= inv;
    int i0 = 0;
#pragma unroll
    for (int e = 1; e < kE; ++e) if (p[e] > p[i0]) i0 = e;   // ties -> low idx
    int i1 = (i0 == 0) ? 1 : 0;
#pragma unroll
    for (int e = 0; e < kE; ++e) if (e != i0 && p[e] > p[i1]) i1 = e;
    if (lane == 0) {
        gates[n * 2 + 0] = p[i0];
        gates[n * 2 + 1] = p[i1];
        eidx [n * 2 + 0] = i0;
        eidx [n * 2 + 1] = i1;
        slots[n * 2 + 0] = atomicAdd(&counts[i0], 1);
        slots[n * 2 + 1] = atomicAdd(&counts[i1], 1);
    }
}

__global__ void offsets_kernel(const int* __restrict__ counts,
                               int* __restrict__ offs) {
    if (threadIdx.x == 0 && blockIdx.x == 0) {
        int run = 0;
        for (int e = 0; e < kE; ++e) {
            offs[e] = run;
            run += (counts[e] + 63) & ~63;            // 64-row aligned regions
        }
        offs[kE] = run;
    }
}

__global__ void init_rows_kernel(int* __restrict__ tokenList) {
    int i = blockIdx.x * blockDim.x + threadIdx.x;
    if (i < ROWS_MAX) tokenList[i] = 0;               // safe gather for pad rows
}

__global__ void scatter_kernel(const int* __restrict__ eidx,
                               const int* __restrict__ slots,
                               const int* __restrict__ offs,
                               int* __restrict__ tokenList,
                               int* __restrict__ rowIdx) {
    int n = blockIdx.x * blockDim.x + threadIdx.x;
    if (n >= kN) return;
#pragma unroll
    for (int k = 0; k < 2; ++k) {
        int e = eidx[n * 2 + k];
        int r = offs[e] + slots[n * 2 + k];
        tokenList[r]      = n;
        rowIdx[n * 2 + k] = r;
    }
}

// ------------------------------ expert GEMMs -------------------------------
// Block: 8 waves; 64-row x 256-col output tile. Every wave owns a 64x32
// slice: 8 accumulators (4 m-subtiles x 2 n-subtiles), so one B-fragment
// pair is reused by 8 WMMAs. A (64 x KT bf16) double-buffered in LDS via
// async copies; row stride 136 ushorts keeps ds_load_b128 conflicts minimal.

__global__ __launch_bounds__(256) void gemm1_kernel(
    const unsigned short* __restrict__ Xb,   // [N][D] bf16
    const unsigned short* __restrict__ W1t,  // [E][H][D] bf16 (B^T)
    const float* __restrict__ b1,            // [E][H]
    const int* __restrict__ counts, const int* __restrict__ offs,
    const int* __restrict__ tokenList,
    unsigned short* __restrict__ Hb)         // [ROWS_MAX][H] bf16
{
    const int e     = blockIdx.y;
    const int mTile = blockIdx.x;
    const int cnt   = counts[e];
    if (mTile * 64 >= cnt) return;                    // uniform early-exit
    const int rowBase = offs[e] + mTile * 64;

    __shared__ unsigned short As[2][64][KT + 8];

    const int tid  = threadIdx.x;
    const int lane = tid & 31;
    const int wave = tid >> 5;
    const int half = lane >> 4;
    const int l16  = lane & 15;
    const int n0   = blockIdx.z * 256 + wave * 32;    // column base in H

    // async staging coords: 64 rows x 16 chunks of 16B, 4 per thread
    const int ar = tid >> 4;                          // rows 0..15 (+16*it)
    const int ac = tid & 15;                          // 16B chunk in row
    int tok[4];
#pragma unroll
    for (int it = 0; it < 4; ++it)
        tok[it] = tokenList[rowBase + ar + it * 16];

    v8f acc[4][2];
#pragma unroll
    for (int m = 0; m < 4; ++m) { acc[m][0] = {}; acc[m][1] = {}; }
    const unsigned short* Bt = W1t + (size_t)e * kH * kD;

    const int nChunks = kD / KT;                      // 8
    // prologue: stage chunk 0 into buffer 0
#pragma unroll
    for (int it = 0; it < 4; ++it)
        async_ld16(&As[0][ar + it * 16][ac * 8],
                   Xb + (size_t)tok[it] * kD + ac * 8);
    for (int ci = 0; ci < nChunks; ++ci) {
        const int cb = ci & 1;
        if (ci + 1 < nChunks) {
#pragma unroll
            for (int it = 0; it < 4; ++it)
                async_ld16(&As[cb ^ 1][ar + it * 16][ac * 8],
                           Xb + (size_t)tok[it] * kD + (ci + 1) * KT + ac * 8);
            wait_async_keep4();                       // chunk ci landed
        } else {
            wait_async_all();
        }
        __syncthreads();
#pragma unroll
        for (int kb = 0; kb < KT; kb += 32) {
            FragBF fb0, fb1;
            // B (contiguous K order, matches A): lane = column, K = 16h..16h+15 contig
            const unsigned short* pb0 =
                Bt + (size_t)(n0 + l16) * kD + ci * KT + kb + half * 16;
            __builtin_prefetch((const void*)(pb0 + 64), 0, 1);
            fb0.q[0] = *(const v4u*)pb0;
            fb0.q[1] = *(const v4u*)(pb0 + 8);
            const unsigned short* pb1 = pb0 + (size_t)16 * kD;
            fb1.q[0] = *(const v4u*)pb1;
            fb1.q[1] = *(const v4u*)(pb1 + 8);
#pragma unroll
            for (int m = 0; m < 4; ++m) {
                FragBF a;
                // A: K = 16h..16h+15 contig (matches B)
                const unsigned short* pa =
                    &As[cb][m * 16 + l16][kb + half * 16];
                a.q[0] = *(const v4u*)pa;
                a.q[1] = *(const v4u*)(pa + 8);
                acc[m][0] = __builtin_amdgcn_wmma_f32_16x16x32_bf16(
                    false, a.v, false, fb0.v, (short)0, acc[m][0], false, false);
                acc[m][1] = __builtin_amdgcn_wmma_f32_16x16x32_bf16(
                    false, a.v, false, fb1.v, (short)0, acc[m][1], false, false);
            }
        }
        __syncthreads();                              // buffer reuse fence
    }
    // C layout: lane<16 -> M=j, lane>=16 -> M=j+8; col = n0 + (lane&15)
    const int col0 = n0 + l16, col1 = col0 + 16;
    const float bias0 = b1[e * kH + col0];
    const float bias1 = b1[e * kH + col1];
#pragma unroll
    for (int m = 0; m < 4; ++m) {
#pragma unroll
        for (int j = 0; j < 8; ++j) {
            int r = rowBase + m * 16 + j + half * 8;
            float v0 = acc[m][0][j] + bias0; v0 = v0 > 0.f ? v0 : 0.f;
            float v1 = acc[m][1][j] + bias1; v1 = v1 > 0.f ? v1 : 0.f;
            Hb[(size_t)r * kH + col0] = f2bf(v0);
            Hb[(size_t)r * kH + col1] = f2bf(v1);
        }
    }
}

__global__ __launch_bounds__(256) void gemm2_kernel(
    const unsigned short* __restrict__ Hb,   // [ROWS_MAX][H] bf16
    const unsigned short* __restrict__ W2t,  // [E][D][H] bf16 (B^T)
    const float* __restrict__ b2,            // [E][D]
    const int* __restrict__ counts, const int* __restrict__ offs,
    float* __restrict__ Y)                   // [ROWS_MAX][D] fp32
{
    const int e     = blockIdx.y;
    const int mTile = blockIdx.x;
    const int cnt   = counts[e];
    if (mTile * 64 >= cnt) return;
    const int rowBase = offs[e] + mTile * 64;

    __shared__ unsigned short As[2][64][KT + 8];

    const int tid  = threadIdx.x;
    const int lane = tid & 31;
    const int wave = tid >> 5;
    const int half = lane >> 4;
    const int l16  = lane & 15;
    const int n0   = blockIdx.z * 256 + wave * 32;    // column base in D

    const int ar = tid >> 4;
    const int ac = tid & 15;

    v8f acc[4][2];
#pragma unroll
    for (int m = 0; m < 4; ++m) { acc[m][0] = {}; acc[m][1] = {}; }
    const unsigned short* Bt = W2t + (size_t)e * kD * kH;

    const int nChunks = kH / KT;                      // 16
#pragma unroll
    for (int it = 0; it < 4; ++it)
        async_ld16(&As[0][ar + it * 16][ac * 8],
                   Hb + (size_t)(rowBase + ar + it * 16) * kH + ac * 8);
    for (int ci = 0; ci < nChunks; ++ci) {
        const int cb = ci & 1;
        if (ci + 1 < nChunks) {
#pragma unroll
            for (int it = 0; it < 4; ++it)
                async_ld16(&As[cb ^ 1][ar + it * 16][ac * 8],
                           Hb + (size_t)(rowBase + ar + it * 16) * kH +
                               (ci + 1) * KT + ac * 8);
            wait_async_keep4();
        } else {
            wait_async_all();
        }
        __syncthreads();
#pragma unroll
        for (int kb = 0; kb < KT; kb += 32) {
            FragBF fb0, fb1;
            const unsigned short* pb0 =
                Bt + (size_t)(n0 + l16) * kH + ci * KT + kb + half * 16;
            __builtin_prefetch((const void*)(pb0 + 64), 0, 1);
            fb0.q[0] = *(const v4u*)pb0;
            fb0.q[1] = *(const v4u*)(pb0 + 8);
            const unsigned short* pb1 = pb0 + (size_t)16 * kH;
            fb1.q[0] = *(const v4u*)pb1;
            fb1.q[1] = *(const v4u*)(pb1 + 8);
#pragma unroll
            for (int m = 0; m < 4; ++m) {
                FragBF a;
                const unsigned short* pa =
                    &As[cb][m * 16 + l16][kb + half * 16];
                a.q[0] = *(const v4u*)pa;
                a.q[1] = *(const v4u*)(pa + 8);
                acc[m][0] = __builtin_amdgcn_wmma_f32_16x16x32_bf16(
                    false, a.v, false, fb0.v, (short)0, acc[m][0], false, false);
                acc[m][1] = __builtin_amdgcn_wmma_f32_16x16x32_bf16(
                    false, a.v, false, fb1.v, (short)0, acc[m][1], false, false);
            }
        }
        __syncthreads();
    }
    const int col0 = n0 + l16, col1 = col0 + 16;
    const float bias0 = b2[e * kD + col0];
    const float bias1 = b2[e * kD + col1];
#pragma unroll
    for (int m = 0; m < 4; ++m) {
#pragma unroll
        for (int j = 0; j < 8; ++j) {
            int r = rowBase + m * 16 + j + half * 8;
            Y[(size_t)r * kD + col0] = acc[m][0][j] + bias0;
            Y[(size_t)r * kD + col1] = acc[m][1][j] + bias1;
        }
    }
}

// ------------------------------ combine ------------------------------------

__global__ __launch_bounds__(256) void combine_kernel(
    const float* __restrict__ Y, const float* __restrict__ gates,
    const int* __restrict__ rowIdx, float* __restrict__ out) {
    constexpr int DV = kD / 4;
    int i = blockIdx.x * 256 + threadIdx.x;
    if (i >= kN * DV) return;
    int n = i / DV, j = i % DV;
    const float4* y0 = (const float4*)(Y + (size_t)rowIdx[n * 2 + 0] * kD);
    const float4* y1 = (const float4*)(Y + (size_t)rowIdx[n * 2 + 1] * kD);
    float g0 = gates[n * 2 + 0], g1 = gates[n * 2 + 1];
    float4 a = y0[j], b = y1[j];
    float4 o;
    o.x = g0 * a.x + g1 * b.x;
    o.y = g0 * a.y + g1 * b.y;
    o.z = g0 * a.z + g1 * b.z;
    o.w = g0 * a.w + g1 * b.w;
    ((float4*)out)[(size_t)n * DV + j] = o;
}

// ------------------------------ launcher -----------------------------------

extern "C" void kernel_launch(void* const* d_in, const int* in_sizes, int n_in,
                              void* d_out, int out_size, void* d_ws,
                              size_t ws_size, hipStream_t stream) {
    const float* x  = (const float*)d_in[0];
    const float* gW = (const float*)d_in[1];
    const float* gb = (const float*)d_in[2];
    const float* W1 = (const float*)d_in[3];
    const float* b1 = (const float*)d_in[4];
    const float* W2 = (const float*)d_in[5];
    const float* b2 = (const float*)d_in[6];
    float* out = (float*)d_out;

    // workspace carve-out (~146 MB total)
    char* p = (char*)d_ws;
    auto take = [&](size_t bytes) -> char* {
        char* r = p;
        p += (bytes + 255) & ~(size_t)255;
        return r;
    };
    unsigned short* Xb  = (unsigned short*)take((size_t)kN * kD * 2);
    unsigned short* W1t = (unsigned short*)take((size_t)kE * kH * kD * 2);
    unsigned short* W2t = (unsigned short*)take((size_t)kE * kD * kH * 2);
    unsigned short* Hb  = (unsigned short*)take((size_t)ROWS_MAX * kH * 2);
    float* Y            = (float*)take((size_t)ROWS_MAX * kD * 4);
    int*   counts       = (int*)take(64);
    int*   offs         = (int*)take(64);
    float* gates        = (float*)take((size_t)kN * 2 * 4);
    int*   eidx         = (int*)take((size_t)kN * 2 * 4);
    int*   slots        = (int*)take((size_t)kN * 2 * 4);
    int*   rowIdx       = (int*)take((size_t)kN * 2 * 4);
    int*   tokenList    = (int*)take((size_t)ROWS_MAX * 4);

    zero_counts_kernel<<<1, 32, 0, stream>>>(counts);
    convert_x_kernel<<<(kN * kD / 4 + 255) / 256, 256, 0, stream>>>(x, Xb);
    transpose_bf16_kernel<<<dim3(kH / 32, kD / 32, kE), 256, 0, stream>>>(
        W1, W1t, kD, kH);  // [E][D][H] -> [E][H][D]
    transpose_bf16_kernel<<<dim3(kD / 32, kH / 32, kE), 256, 0, stream>>>(
        W2, W2t, kH, kD);  // [E][H][D] -> [E][D][H]
    gate_kernel<<<kN / 8, 256, 0, stream>>>(x, gW, gb, gates, eidx, slots,
                                            counts);
    offsets_kernel<<<1, 32, 0, stream>>>(counts, offs);
    init_rows_kernel<<<(ROWS_MAX + 255) / 256, 256, 0, stream>>>(tokenList);
    scatter_kernel<<<(kN + 255) / 256, 256, 0, stream>>>(eidx, slots, offs,
                                                         tokenList, rowIdx);
    gemm1_kernel<<<dim3(kN / 64, kE, kH / 256), 256, 0, stream>>>(
        Xb, W1t, b1, counts, offs, tokenList, Hb);
    gemm2_kernel<<<dim3(kN / 64, kE, kD / 256), 256, 0, stream>>>(
        Hb, W2t, b2, counts, offs, Y);
    combine_kernel<<<(kN * (kD / 4) + 255) / 256, 256, 0, stream>>>(
        Y, gates, rowIdx, out);
}
